// RdSelfAttention_53730040873211
// MI455X (gfx1250) — matrix operand order repaired and checked by reading, not compiled, not run on hardware
//
#include <hip/hip_runtime.h>
#include <hip/hip_bf16.h>
#include <math.h>

typedef _Float16 half_t;
typedef __attribute__((ext_vector_type(16))) _Float16 v16h;
typedef __attribute__((ext_vector_type(8)))  _Float16 v8h;
typedef __attribute__((ext_vector_type(8)))  float    v8f;

#define HEADS 8
#define MID   32
#define CIN   64
#define COUT  64
#define WDIM  56
#define HW    3136   // 56*56

static __device__ __forceinline__ v16h make_v16(v8h lo, v8h hi) {
  v16h r;
#pragma unroll
  for (int i = 0; i < 8; ++i) { r[i] = lo[i]; r[8 + i] = hi[i]; }
  return r;
}

// ---------------------------------------------------------------------------
// K0: convert weights to f16 (fold 1/sqrt(CIN) into Wq/bq)
// ---------------------------------------------------------------------------
__global__ void prep_weights(const float* __restrict__ Wq, const float* __restrict__ bq,
                             const float* __restrict__ Wk, const float* __restrict__ Wv,
                             const float* __restrict__ Wo,
                             half_t* __restrict__ Wq16, half_t* __restrict__ Wk16,
                             half_t* __restrict__ Wv16, half_t* __restrict__ Wo16,
                             float* __restrict__ bqs, float inv) {
  const int n = 256 * 64;
  for (int t = blockIdx.x * blockDim.x + threadIdx.x; t < n; t += gridDim.x * blockDim.x) {
    Wq16[t] = (half_t)(Wq[t] * inv);
    Wk16[t] = (half_t)Wk[t];
    Wv16[t] = (half_t)Wv[t];
    Wo16[t] = (half_t)Wo[t];
    if (t < 256) bqs[t] = bq[t] * inv;
  }
}

// ---------------------------------------------------------------------------
// K1: QKV projection via WMMA. One wave per 16-pixel tile.
//   q,k stored per head as [pixel][32ch] f16 ; v stored per head [32ch][pixel]
// ---------------------------------------------------------------------------
__global__ void __launch_bounds__(32) proj_qkv(
    const float* __restrict__ x,
    const half_t* __restrict__ Wq16, const float* __restrict__ bqs,
    const half_t* __restrict__ Wk16, const float* __restrict__ bk,
    const half_t* __restrict__ Wv16, const float* __restrict__ bv,
    half_t* __restrict__ qh, half_t* __restrict__ kh, half_t* __restrict__ vh) {
  __shared__ half_t xt[16 * 64];          // [pixel][cin] f16
  const int lane = threadIdx.x;
  const int l15 = lane & 15, hi = lane >> 4;
  const int p0 = blockIdx.x * 16;

  // stage x tile transposed into LDS: xt[p][c]
#pragma unroll
  for (int g = 0; g < 2; ++g) {
    const int c = lane + g * 32;
    const float4* src = (const float4*)(x + (size_t)c * HW + p0);
    float4 f0 = src[0], f1 = src[1], f2 = src[2], f3 = src[3];
    float vv[16] = {f0.x, f0.y, f0.z, f0.w, f1.x, f1.y, f1.z, f1.w,
                    f2.x, f2.y, f2.z, f2.w, f3.x, f3.y, f3.z, f3.w};
#pragma unroll
    for (int p = 0; p < 16; ++p) xt[p * 64 + c] = (half_t)vv[p];
  }
  asm volatile("s_wait_dscnt 0" ::: "memory");

  // A fragments for the two K-blocks of cin=64
  v16h ax[2];
#pragma unroll
  for (int kb = 0; kb < 2; ++kb) {
    const half_t* rp = xt + l15 * 64 + kb * 32 + hi * 8;
    ax[kb] = make_v16(*(const v8h*)rp, *(const v8h*)(rp + 16));
  }

  const half_t* Ws[3] = {Wq16, Wk16, Wv16};
  const float*  Bs[3] = {bqs, bk, bv};
#pragma unroll
  for (int mat = 0; mat < 3; ++mat) {
    const half_t* W  = Ws[mat];
    const float*  bb = Bs[mat];
    for (int dt = 0; dt < 16; ++dt) {
      const int d = dt * 16 + l15;
      const float bval = bb[d];
      v8f c = {};
#pragma unroll
      for (int kb = 0; kb < 2; ++kb) {
        const half_t* wp = W + (size_t)d * 64 + kb * 32 + hi * 8;
        v16h bf = make_v16(*(const v8h*)wp, *(const v8h*)(wp + 16));
        c = __builtin_amdgcn_wmma_f32_16x16x32_f16(false, ax[kb], false, bf,
                                                   (short)0, c, false, false);
      }
      const int mid = d >> 3, hd = d & 7;  // channel d = mid*HEADS + hd
#pragma unroll
      for (int r = 0; r < 8; ++r) {
        const int pix = p0 + r + 8 * hi;
        const half_t h = (half_t)(c[r] + bval);
        if (mat == 0)      qh[((size_t)hd * HW + pix) * MID + mid] = h;
        else if (mat == 1) kh[((size_t)hd * HW + pix) * MID + mid] = h;
        else               vh[((size_t)hd * MID + mid) * HW + pix] = h;
      }
    }
  }
}

// ---------------------------------------------------------------------------
// K2: relative positional bias tables.
//   rowS[hd][p][k] = sum_{c<16}  q[hd][p][c]    * row_tab[k - i + 55][c]
//   colS[hd][p][l] = sum_{c<16}  q[hd][p][16+c] * col_tab[l - j + 55][c]
// (q already carries the 1/sqrt(CIN) scale)
// ---------------------------------------------------------------------------
__global__ void pos_bias(const half_t* __restrict__ qh,
                         const float* __restrict__ rowT, const float* __restrict__ colT,
                         float* __restrict__ rowS, float* __restrict__ colS) {
  const int idx = blockIdx.x * blockDim.x + threadIdx.x;
  if (idx >= HEADS * HW * WDIM) return;
  const int k  = idx % WDIM;
  const int p  = (idx / WDIM) % HW;
  const int hd = idx / (WDIM * HW);
  const int i = p / WDIM, j = p % WDIM;
  const half_t* q = qh + ((size_t)hd * HW + p) * MID;
  const float* re = rowT + (size_t)(k - i + WDIM - 1) * 16;
  const float* ce = colT + (size_t)(k - j + WDIM - 1) * 16;
  float rs = 0.f, cs = 0.f;
#pragma unroll
  for (int c = 0; c < 16; ++c) {
    rs += (float)q[c] * re[c];
    cs += (float)q[16 + c] * ce[c];
  }
  rowS[idx] = rs;
  colS[idx] = cs;
}

// ---------------------------------------------------------------------------
// K3: fused flash attention. One wave per (head, 16-query tile).
//   Streams keys in 32-chunks: 2 WMMA (scores) + online softmax + 2 WMMA (P*V).
//   Output o stored f16 as [pixel][256] with d = mid*8 + hd.
// ---------------------------------------------------------------------------
__global__ void __launch_bounds__(32) attention(
    const half_t* __restrict__ qh, const half_t* __restrict__ kh,
    const half_t* __restrict__ vh,
    const float* __restrict__ rowS, const float* __restrict__ colS,
    half_t* __restrict__ oh) {
  __shared__ half_t pt[16 * 32];          // P tile, row-major [qrow][key32] f16
  const int lane = threadIdx.x;
  const int l15 = lane & 15, hi = lane >> 4;
  const int p0 = blockIdx.x * 16;
  const int hd = blockIdx.y;

  const half_t* qbase = qh + (size_t)hd * HW * MID;
  const half_t* kbase = kh + (size_t)hd * HW * MID;
  const half_t* vbase = vh + (size_t)hd * MID * HW;
  const float*  rbase = rowS + (size_t)hd * HW * WDIM;
  const float*  cbase = colS + (size_t)hd * HW * WDIM;

  // Q A-fragment (16 queries x 32 ch)
  const half_t* qp = qbase + (size_t)(p0 + l15) * MID + hi * 8;
  const v16h aq = make_v16(*(const v8h*)qp, *(const v8h*)(qp + 16));

  float m[8], l[8];
  v8f o0 = {}, o1 = {};
#pragma unroll
  for (int r = 0; r < 8; ++r) { m[r] = -1e30f; l[r] = 0.f; }

  for (int kt = 0; kt < HW / 32; ++kt) {
    const int n0 = kt * 32;
    v8f s[2];
    // scores: S = Q * K^T  (B-frag rows are contiguous [key][32ch])
#pragma unroll
    for (int f = 0; f < 2; ++f) {
      const half_t* kp = kbase + (size_t)(n0 + f * 16 + l15) * MID + hi * 8;
      v16h bk = make_v16(*(const v8h*)kp, *(const v8h*)(kp + 16));
      v8f z = {};
      s[f] = __builtin_amdgcn_wmma_f32_16x16x32_f16(false, aq, false, bk,
                                                    (short)0, z, false, false);
    }
    // add relative-position bias (already 1/sqrt(CIN)-scaled)
#pragma unroll
    for (int f = 0; f < 2; ++f) {
      const int key = n0 + f * 16 + l15;
      const int kr = key / WDIM, kc = key % WDIM;
#pragma unroll
      for (int r = 0; r < 8; ++r) {
        const int qpix = p0 + r + 8 * hi;
        s[f][r] += rbase[(size_t)qpix * WDIM + kr] + cbase[(size_t)qpix * WDIM + kc];
      }
    }
    // online softmax (row = C-layout row owned by 16-lane half-wave)
#pragma unroll
    for (int r = 0; r < 8; ++r) {
      float t = fmaxf(s[0][r], s[1][r]);
#pragma unroll
      for (int off = 8; off > 0; off >>= 1) t = fmaxf(t, __shfl_xor(t, off, 16));
      const float mn = fmaxf(m[r], t);
      const float sc = __expf(m[r] - mn);
      m[r] = mn;
      const float e0 = __expf(s[0][r] - mn);
      const float e1 = __expf(s[1][r] - mn);
      s[0][r] = e0; s[1][r] = e1;
      float rs = e0 + e1;
#pragma unroll
      for (int off = 8; off > 0; off >>= 1) rs += __shfl_xor(rs, off, 16);
      l[r] = l[r] * sc + rs;
      o0[r] *= sc; o1[r] *= sc;
    }
    // C-layout -> A-fragment transpose through LDS (single wave: DS in-order)
#pragma unroll
    for (int f = 0; f < 2; ++f)
#pragma unroll
      for (int r = 0; r < 8; ++r)
        pt[(r + 8 * hi) * 32 + f * 16 + l15] = (half_t)s[f][r];
    asm volatile("s_wait_dscnt 0" ::: "memory");
    const half_t* pp = pt + l15 * 32 + hi * 8;
    const v16h ap = make_v16(*(const v8h*)pp, *(const v8h*)(pp + 16));
    asm volatile("s_wait_dscnt 0" ::: "memory");
    // O += P * V  (V channel-major: key pairs contiguous per channel row)
#pragma unroll
    for (int f = 0; f < 2; ++f) {
      const half_t* vp = vbase + (size_t)(f * 16 + l15) * HW + n0 + hi * 8;
      v16h bv = make_v16(*(const v8h*)vp, *(const v8h*)(vp + 16));
      if (f == 0)
        o0 = __builtin_amdgcn_wmma_f32_16x16x32_f16(false, ap, false, bv,
                                                    (short)0, o0, false, false);
      else
        o1 = __builtin_amdgcn_wmma_f32_16x16x32_f16(false, ap, false, bv,
                                                    (short)0, o1, false, false);
    }
  }
  // normalize and store o as f16 in [pixel][256], d = mid*8 + hd
#pragma unroll
  for (int r = 0; r < 8; ++r) {
    const float rinv = 1.0f / l[r];
    const int pix = p0 + r + 8 * hi;
    oh[(size_t)pix * 256 + (size_t)l15 * 8 + hd]        = (half_t)(o0[r] * rinv);
    oh[(size_t)pix * 256 + (size_t)(l15 + 16) * 8 + hd] = (half_t)(o1[r] * rinv);
  }
}

// ---------------------------------------------------------------------------
// K4: output projection [3136x256] x Wo^T[256x64] via WMMA. One wave / 16 px.
// ---------------------------------------------------------------------------
__global__ void __launch_bounds__(32) out_proj(
    const half_t* __restrict__ oh, const half_t* __restrict__ Wo16,
    const float* __restrict__ bo, float* __restrict__ out) {
  const int lane = threadIdx.x;
  const int l15 = lane & 15, hi = lane >> 4;
  const int p0 = blockIdx.x * 16;

  v16h af[8];
#pragma unroll
  for (int kb = 0; kb < 8; ++kb) {
    const half_t* ap = oh + (size_t)(p0 + l15) * 256 + kb * 32 + hi * 8;
    af[kb] = make_v16(*(const v8h*)ap, *(const v8h*)(ap + 16));
  }
#pragma unroll
  for (int nt = 0; nt < 4; ++nt) {
    const int cout = nt * 16 + l15;
    const float bval = bo[cout];
    v8f c = {};
#pragma unroll
    for (int kb = 0; kb < 8; ++kb) {
      const half_t* wp = Wo16 + (size_t)cout * 256 + kb * 32 + hi * 8;
      v16h bf = make_v16(*(const v8h*)wp, *(const v8h*)(wp + 16));
      c = __builtin_amdgcn_wmma_f32_16x16x32_f16(false, af[kb], false, bf,
                                                 (short)0, c, false, false);
    }
#pragma unroll
    for (int r = 0; r < 8; ++r) {
      const int pix = p0 + r + 8 * hi;
      out[(size_t)cout * HW + pix] = c[r] + bval;
    }
  }
}

// ---------------------------------------------------------------------------
extern "C" void kernel_launch(void* const* d_in, const int* in_sizes, int n_in,
                              void* d_out, int out_size, void* d_ws, size_t ws_size,
                              hipStream_t stream) {
  (void)in_sizes; (void)n_in; (void)out_size; (void)ws_size;
  const float* x  = (const float*)d_in[0];
  const float* Wq = (const float*)d_in[1];
  const float* bq = (const float*)d_in[2];
  const float* Wk = (const float*)d_in[3];
  const float* bk = (const float*)d_in[4];
  const float* Wv = (const float*)d_in[5];
  const float* bv = (const float*)d_in[6];
  const float* Wo = (const float*)d_in[7];
  const float* bo = (const float*)d_in[8];
  const float* rT = (const float*)d_in[9];
  const float* cT = (const float*)d_in[10];
  float* out = (float*)d_out;

  char* ws = (char*)d_ws;
  size_t off = 0;
  auto alloc = [&](size_t bytes) -> void* {
    void* p = ws + off;
    off = (off + bytes + 255) & ~(size_t)255;
    return p;
  };
  half_t* Wq16 = (half_t*)alloc(256 * 64 * sizeof(half_t));
  half_t* Wk16 = (half_t*)alloc(256 * 64 * sizeof(half_t));
  half_t* Wv16 = (half_t*)alloc(256 * 64 * sizeof(half_t));
  half_t* Wo16 = (half_t*)alloc(64 * 256 * sizeof(half_t));
  float*  bqs  = (float*) alloc(256 * sizeof(float));
  half_t* qh   = (half_t*)alloc((size_t)HEADS * HW * MID * sizeof(half_t));
  half_t* kh   = (half_t*)alloc((size_t)HEADS * HW * MID * sizeof(half_t));
  half_t* vh   = (half_t*)alloc((size_t)HEADS * MID * HW * sizeof(half_t));
  half_t* oh   = (half_t*)alloc((size_t)HW * 256 * sizeof(half_t));
  float*  rowS = (float*) alloc((size_t)HEADS * HW * WDIM * sizeof(float));
  float*  colS = (float*) alloc((size_t)HEADS * HW * WDIM * sizeof(float));

  const float inv = 0.125f;  // 1/sqrt(64)

  prep_weights<<<64, 256, 0, stream>>>(Wq, bq, Wk, Wv, Wo,
                                       Wq16, Wk16, Wv16, Wo16, bqs, inv);
  proj_qkv<<<HW / 16, 32, 0, stream>>>(x, Wq16, bqs, Wk16, bk, Wv16, bv, qh, kh, vh);
  const int nb = HEADS * HW * WDIM;
  pos_bias<<<(nb + 255) / 256, 256, 0, stream>>>(qh, rT, cT, rowS, colS);
  attention<<<dim3(HW / 16, HEADS), 32, 0, stream>>>(qh, kh, vh, rowS, colS, oh);
  out_proj<<<HW / 16, 32, 0, stream>>>(oh, Wo16, bo, out);
}
